// MHA_29832842838282
// MI455X (gfx1250) — matrix operand order repaired and checked by reading, not compiled, not run on hardware
//
#include <hip/hip_runtime.h>

// ---------------------------------------------------------------------------
// MHA forward for MI455X (gfx1250, wave32, WMMA bf16 16x16x32, f32 accum).
// B=8, T=1024, D=1024, H=16, dh=64.
// Workspace layout (requires 72 MB):
//   [ 0MB..16MB)  x_bf16 [8192 x 1024]  -- reused as ctx after projections
//   [16MB..18MB)  Wq^T bf16 [1024 x 1024]
//   [18MB..20MB)  Wk^T
//   [20MB..22MB)  Wv^T
//   [22MB..24MB)  Wp^T
//   [24MB..40MB)  Q  bf16 [B,H,T,64]
//   [40MB..56MB)  K  bf16 [B,H,T,64]
//   [56MB..72MB)  V^T bf16 [B,H,64,T]
// ---------------------------------------------------------------------------

typedef __attribute__((ext_vector_type(16))) __bf16 v16bf;
typedef __attribute__((ext_vector_type(8)))  float  v8f;

struct alignas(16) U4 { unsigned int a, b, c, d; };
struct alignas(32) U8 { U4 lo, hi; };
struct alignas(16) F4 { float x, y, z, w; };
struct alignas(8)  US4 { unsigned short x, y, z, w; };

static __device__ __forceinline__ v16bf make_frag(U4 lo, U4 hi) {
  U8 u{lo, hi};
  return __builtin_bit_cast(v16bf, u);
}

// f32 -> bf16 bits, round-to-nearest-even
static __device__ __forceinline__ unsigned short f2bf(float f) {
  unsigned u = __builtin_bit_cast(unsigned, f);
  u += 0x7fffu + ((u >> 16) & 1u);
  return (unsigned short)(u >> 16);
}

// A-fragment: 16x32 bf16 tile, row-major source with leading dim `ld`.
// lane (lrow = lane&15, lhalf = lane>>4): row m = lrow,
// K = lhalf*16 + {0..15}, contiguous (matches load_b)
static __device__ __forceinline__ v16bf load_a(const unsigned short* base, int ld,
                                               int lrow, int lhalf) {
  const unsigned short* p = base + lrow * ld + lhalf * 16;
  U4 lo = *reinterpret_cast<const U4*>(p);
  U4 hi = *reinterpret_cast<const U4*>(p + 8);
  return make_frag(lo, hi);
}

// B-fragment for B (K x N), supplied as B^T (N x K row-major, leading dim ld).
// lane: col n = lrow, K = lhalf*16 + {0..15}, contiguous (same order as load_a)
static __device__ __forceinline__ v16bf load_b(const unsigned short* baseT, int ld,
                                               int lrow, int lhalf) {
  const unsigned short* p = baseT + lrow * ld + lhalf * 16;
  U4 lo = *reinterpret_cast<const U4*>(p);
  U4 hi = *reinterpret_cast<const U4*>(p + 8);
  return make_frag(lo, hi);
}

static __device__ __forceinline__ v8f wmma_bf16(v16bf a, v16bf b, v8f c) {
  // D = A*B + C, f32 accumulate
  return __builtin_amdgcn_wmma_f32_16x16x32_bf16(false, a, false, b, (short)0, c,
                                                 false, false);
}

static __device__ __forceinline__ v8f zero8() {
  v8f z;
#pragma unroll
  for (int i = 0; i < 8; ++i) z[i] = 0.0f;
  return z;
}

// ---------------------------------------------------------------------------
// x (f32) -> bf16, vectorized 4 at a time
// ---------------------------------------------------------------------------
__global__ __launch_bounds__(256) void cvt_x_kernel(const float* __restrict__ src,
                                                    unsigned short* __restrict__ dst,
                                                    int n) {
  int i = (blockIdx.x * 256 + threadIdx.x) * 4;
  if (i < n) {
    F4 f = *reinterpret_cast<const F4*>(src + i);
    US4 u{f2bf(f.x), f2bf(f.y), f2bf(f.z), f2bf(f.w)};
    *reinterpret_cast<US4*>(dst + i) = u;
  }
}

// ---------------------------------------------------------------------------
// W [1024 x 1024] f32 -> W^T bf16, LDS-tiled 32x32 transpose.
// block = 256 (32x8), grid = (32, 32). Coalesced reads AND writes.
// ---------------------------------------------------------------------------
__global__ __launch_bounds__(256) void transpose_w_kernel(const float* __restrict__ W,
                                                          unsigned short* __restrict__ Wt) {
  __shared__ float tile[32][33];
  const int tx = threadIdx.x & 31;
  const int ty = threadIdx.x >> 5;  // 0..7
  const int k0 = blockIdx.y * 32;
  const int n0 = blockIdx.x * 32;
#pragma unroll
  for (int i = 0; i < 32; i += 8)
    tile[ty + i][tx] = W[(size_t)(k0 + ty + i) * 1024 + n0 + tx];
  __syncthreads();
#pragma unroll
  for (int i = 0; i < 32; i += 8)
    Wt[(size_t)(n0 + ty + i) * 1024 + k0 + tx] = f2bf(tile[tx][ty + i]);
}

// ---------------------------------------------------------------------------
// Wave GEMM helpers: one k-step fragment set (32x64 tile -> 2 A + 4 B frags)
// ---------------------------------------------------------------------------
struct Frags {
  v16bf a0, a1, b0, b1, b2, b3;
};

static __device__ __forceinline__ void load_frags(Frags& f, const unsigned short* A0,
                                                  const unsigned short* A1,
                                                  const unsigned short* Bp, int k,
                                                  int lrow, int lhalf) {
  f.a0 = load_a(A0 + k, 1024, lrow, lhalf);
  f.a1 = load_a(A1 + k, 1024, lrow, lhalf);
  f.b0 = load_b(Bp + 0 * 16 * 1024 + k, 1024, lrow, lhalf);
  f.b1 = load_b(Bp + 1 * 16 * 1024 + k, 1024, lrow, lhalf);
  f.b2 = load_b(Bp + 2 * 16 * 1024 + k, 1024, lrow, lhalf);
  f.b3 = load_b(Bp + 3 * 16 * 1024 + k, 1024, lrow, lhalf);
}

static __device__ __forceinline__ void do_wmmas(v8f acc[2][4], const Frags& f) {
  acc[0][0] = wmma_bf16(f.a0, f.b0, acc[0][0]);
  acc[0][1] = wmma_bf16(f.a0, f.b1, acc[0][1]);
  acc[0][2] = wmma_bf16(f.a0, f.b2, acc[0][2]);
  acc[0][3] = wmma_bf16(f.a0, f.b3, acc[0][3]);
  acc[1][0] = wmma_bf16(f.a1, f.b0, acc[1][0]);
  acc[1][1] = wmma_bf16(f.a1, f.b1, acc[1][1]);
  acc[1][2] = wmma_bf16(f.a1, f.b2, acc[1][2]);
  acc[1][3] = wmma_bf16(f.a1, f.b3, acc[1][3]);
}

// ---------------------------------------------------------------------------
// Wave GEMM: C[8192 x 1024] = A(bf16, row-major, ld=1024) * B, with B given as
// B^T (bf16, 1024 x 1024 row-major). One wave -> 32x64 output tile.
// Ping-pong double-buffered fragments (no register rotation, no copies).
// grid = (N/64=16, M/32/8=32), block = 256 (8 waves).
// mode 0: f32 row-major store to outF (final projection)
// mode 1: bf16 store to [b,h,t,64]          (Q, K)
// mode 2: bf16 store to [b,h,64,t]          (V transposed)
// ---------------------------------------------------------------------------
__global__ __launch_bounds__(256) void gemm_32x64_kernel(
    const unsigned short* __restrict__ A, const unsigned short* __restrict__ BT,
    float* __restrict__ outF, unsigned short* __restrict__ outB, int mode) {
  const int lane  = threadIdx.x & 31;
  const int wave  = threadIdx.x >> 5;
  const int lrow  = lane & 15;
  const int lhalf = lane >> 4;
  const int mtile = blockIdx.y * 8 + wave;  // 0..255 (32-row tiles)
  const int ntile = blockIdx.x;             // 0..15  (64-col tiles)

  const unsigned short* A0 = A + (size_t)mtile * 32 * 1024;
  const unsigned short* A1 = A0 + 16 * 1024;
  const unsigned short* Bp = BT + (size_t)ntile * 64 * 1024;

  v8f acc[2][4];
#pragma unroll
  for (int mt = 0; mt < 2; ++mt)
#pragma unroll
    for (int j = 0; j < 4; ++j) acc[mt][j] = zero8();

  Frags f0, f1;
  load_frags(f0, A0, A1, Bp, 0, lrow, lhalf);
  load_frags(f1, A0, A1, Bp, 32, lrow, lhalf);

  // steady state: consume one buffer, immediately refill it 64 elements ahead
  for (int k = 0; k < 1024 - 64; k += 64) {
    do_wmmas(acc, f0);
    load_frags(f0, A0, A1, Bp, k + 64, lrow, lhalf);
    do_wmmas(acc, f1);
    load_frags(f1, A0, A1, Bp, k + 96, lrow, lhalf);
  }
  // epilogue: f0 holds k=960, f1 holds k=992
  do_wmmas(acc, f0);
  do_wmmas(acc, f1);

#pragma unroll
  for (int mt = 0; mt < 2; ++mt) {
#pragma unroll
    for (int v = 0; v < 8; ++v) {
      int m = mtile * 32 + mt * 16 + v + 8 * lhalf;  // global row (b*1024 + t)
#pragma unroll
      for (int j = 0; j < 4; ++j) {
        int col = ntile * 64 + j * 16 + lrow;  // global col (h*64 + jd)
        float val = acc[mt][j][v];
        if (mode == 0) {
          outF[(size_t)m * 1024 + col] = val;
        } else {
          int bb = m >> 10, t = m & 1023, h = col >> 6, jd = col & 63;
          if (mode == 1)
            outB[(((size_t)bb * 16 + h) * 1024 + t) * 64 + jd] = f2bf(val);
          else
            outB[(((size_t)bb * 16 + h) * 64 + jd) * 1024 + t] = f2bf(val);
        }
      }
    }
  }
}

// ---------------------------------------------------------------------------
// Flash attention helpers
// ---------------------------------------------------------------------------
struct KFrags {
  v16bf k00, k01, k10, k11;  // 32-key tile: [cols 0-15 | 16-31] x [dh 0-31 | 32-63]
};

static __device__ __forceinline__ KFrags load_kfrags(const unsigned short* Kt0,
                                                     int lrow, int lhalf) {
  KFrags f;
  f.k00 = load_b(Kt0, 64, lrow, lhalf);
  f.k01 = load_b(Kt0 + 32, 64, lrow, lhalf);
  f.k10 = load_b(Kt0 + 16 * 64, 64, lrow, lhalf);
  f.k11 = load_b(Kt0 + 16 * 64 + 32, 64, lrow, lhalf);
  return f;
}

// One 32-key step: scores, prefetch next K tile, online softmax, P*V.
static __device__ __forceinline__ void attn_step(
    const v16bf& aq0, const v16bf& aq1, const KFrags& kf, KFrags& kf_next,
    const unsigned short* __restrict__ Kb, int kt_next,
    const unsigned short* __restrict__ Vb, int kt,
    unsigned short* Pld, v8f& o0, v8f& o1, v8f& o2, v8f& o3,
    float* mi, float* li, int lrow, int lhalf) {
  // S = Q * K^T for this 16x32 key tile
  v8f s0 = wmma_bf16(aq0, kf.k00, zero8());
  s0     = wmma_bf16(aq1, kf.k01, s0);
  v8f s1 = wmma_bf16(aq0, kf.k10, zero8());
  s1     = wmma_bf16(aq1, kf.k11, s1);

  // prefetch next key tile; latency hidden behind softmax + LDS restage
  kf_next = load_kfrags(Kb + (size_t)kt_next * 32 * 64, lrow, lhalf);

  // issue V fragment loads now as well
  const unsigned short* Vk = Vb + (size_t)kt * 32;
  v16bf vb0 = load_b(Vk + (size_t)0 * 16 * 1024, 1024, lrow, lhalf);
  v16bf vb1 = load_b(Vk + (size_t)1 * 16 * 1024, 1024, lrow, lhalf);
  v16bf vb2 = load_b(Vk + (size_t)2 * 16 * 1024, 1024, lrow, lhalf);
  v16bf vb3 = load_b(Vk + (size_t)3 * 16 * 1024, 1024, lrow, lhalf);

  // online softmax; C-frag row (v + 8*lhalf) lives across the 16 lanes of a half
  const float sc = 0.125f;  // 1/sqrt(dh=64)
  float p0[8], p1[8], corr[8];
#pragma unroll
  for (int v = 0; v < 8; ++v) {
    float x0 = s0[v] * sc, x1 = s1[v] * sc;
    float t = fmaxf(x0, x1);
    t = fmaxf(t, __shfl_xor(t, 1, 32));
    t = fmaxf(t, __shfl_xor(t, 2, 32));
    t = fmaxf(t, __shfl_xor(t, 4, 32));
    t = fmaxf(t, __shfl_xor(t, 8, 32));
    float mnew = fmaxf(mi[v], t);
    float c  = __expf(mi[v] - mnew);
    float e0 = __expf(x0 - mnew);
    float e1 = __expf(x1 - mnew);
    float rs = e0 + e1;
    rs += __shfl_xor(rs, 1, 32);
    rs += __shfl_xor(rs, 2, 32);
    rs += __shfl_xor(rs, 4, 32);
    rs += __shfl_xor(rs, 8, 32);
    li[v] = li[v] * c + rs;
    mi[v] = mnew;
    corr[v] = c;
    p0[v] = e0;
    p1[v] = e1;
  }
#pragma unroll
  for (int v = 0; v < 8; ++v) {
    o0[v] *= corr[v]; o1[v] *= corr[v]; o2[v] *= corr[v]; o3[v] *= corr[v];
  }

  // restage P (16x32) through LDS: C-fragment layout -> A-fragment layout
  __syncthreads();
#pragma unroll
  for (int v = 0; v < 8; ++v) {
    int r = v + 8 * lhalf;
    Pld[r * 32 + lrow]      = f2bf(p0[v]);
    Pld[r * 32 + 16 + lrow] = f2bf(p1[v]);
  }
  __syncthreads();
  v16bf ap = load_a(Pld, 32, lrow, lhalf);

  // O += P * V using the preloaded V fragments
  o0 = wmma_bf16(ap, vb0, o0);
  o1 = wmma_bf16(ap, vb1, o1);
  o2 = wmma_bf16(ap, vb2, o2);
  o3 = wmma_bf16(ap, vb3, o3);
}

// ---------------------------------------------------------------------------
// Flash attention: one wave per (bh, 16-query tile). Streams 32-key tiles with
// ping-pong K-fragment prefetch (unrolled by 2, no register rotation).
// Q,K: [b,h,1024,64] bf16 ; Vt: [b,h,64,1024] bf16 ; ctx out: [8192,1024] bf16
// grid = (T/16=64, B*H=128), block = 32.
// ---------------------------------------------------------------------------
__global__ __launch_bounds__(32) void flash_attn_kernel(
    const unsigned short* __restrict__ Q, const unsigned short* __restrict__ K,
    const unsigned short* __restrict__ Vt, unsigned short* __restrict__ ctx) {
  __shared__ alignas(16) unsigned short Pld[16 * 32];

  const int lane  = threadIdx.x;
  const int lrow  = lane & 15;
  const int lhalf = lane >> 4;
  const int qt = blockIdx.x;  // 0..63
  const int bh = blockIdx.y;  // 0..127

  const unsigned short* Qb = Q + (size_t)bh * 1024 * 64 + (size_t)qt * 16 * 64;
  const unsigned short* Kb = K + (size_t)bh * 1024 * 64;
  const unsigned short* Vb = Vt + (size_t)bh * 64 * 1024;

  const v16bf aq0 = load_a(Qb, 64, lrow, lhalf);       // dh 0..31
  const v16bf aq1 = load_a(Qb + 32, 64, lrow, lhalf);  // dh 32..63

  v8f o0 = zero8(), o1 = zero8(), o2 = zero8(), o3 = zero8();
  float mi[8], li[8];
#pragma unroll
  for (int v = 0; v < 8; ++v) { mi[v] = -1e30f; li[v] = 0.0f; }

  KFrags kfA = load_kfrags(Kb, lrow, lhalf);
  KFrags kfB;

  for (int kt = 0; kt < 64; kt += 2) {
    // even step: consume kfA, prefetch kfB @ kt+1
    attn_step(aq0, aq1, kfA, kfB, Kb, kt + 1, Vb, kt, Pld, o0, o1, o2, o3, mi, li,
              lrow, lhalf);
    // odd step: consume kfB, prefetch kfA @ kt+2 (clamped; last prefetch is dead)
    int ktn = (kt + 2 < 64) ? (kt + 2) : 0;
    attn_step(aq0, aq1, kfB, kfA, Kb, ktn, Vb, kt + 1, Pld, o0, o1, o2, o3, mi, li,
              lrow, lhalf);
  }

  // normalize and write ctx row-major [8192, 1024] bf16
  const int b = bh >> 4, h = bh & 15;
#pragma unroll
  for (int v = 0; v < 8; ++v) {
    float inv = 1.0f / li[v];
    int t = qt * 16 + v + 8 * lhalf;
    size_t rowbase = ((size_t)b * 1024 + t) * 1024 + (size_t)h * 64;
    ctx[rowbase + 0 * 16 + lrow] = f2bf(o0[v] * inv);
    ctx[rowbase + 1 * 16 + lrow] = f2bf(o1[v] * inv);
    ctx[rowbase + 2 * 16 + lrow] = f2bf(o2[v] * inv);
    ctx[rowbase + 3 * 16 + lrow] = f2bf(o3[v] * inv);
  }
}

// ---------------------------------------------------------------------------
extern "C" void kernel_launch(void* const* d_in, const int* in_sizes, int n_in,
                              void* d_out, int out_size, void* d_ws, size_t ws_size,
                              hipStream_t stream) {
  (void)in_sizes; (void)n_in; (void)out_size; (void)ws_size;

  const float* x  = (const float*)d_in[0];
  const float* Wq = (const float*)d_in[1];
  const float* Wk = (const float*)d_in[2];
  const float* Wv = (const float*)d_in[3];
  const float* Wp = (const float*)d_in[4];
  float* out = (float*)d_out;

  char* ws = (char*)d_ws;
  const size_t MB = 1u << 20;
  unsigned short* xb  = (unsigned short*)(ws);             // 16 MB, reused as ctx
  unsigned short* Wqt = (unsigned short*)(ws + 16 * MB);   // 2 MB each
  unsigned short* Wkt = (unsigned short*)(ws + 18 * MB);
  unsigned short* Wvt = (unsigned short*)(ws + 20 * MB);
  unsigned short* Wpt = (unsigned short*)(ws + 22 * MB);
  unsigned short* Qh  = (unsigned short*)(ws + 24 * MB);   // 16 MB
  unsigned short* Kh  = (unsigned short*)(ws + 40 * MB);   // 16 MB
  unsigned short* Vth = (unsigned short*)(ws + 56 * MB);   // 16 MB

  const int NX = 8 * 1024 * 1024;  // elements of x

  // 1) convert inputs to bf16 (weights transposed for contiguous B-fragments)
  cvt_x_kernel<<<NX / (256 * 4), 256, 0, stream>>>(x, xb, NX);
  transpose_w_kernel<<<dim3(32, 32), 256, 0, stream>>>(Wq, Wqt);
  transpose_w_kernel<<<dim3(32, 32), 256, 0, stream>>>(Wk, Wkt);
  transpose_w_kernel<<<dim3(32, 32), 256, 0, stream>>>(Wv, Wvt);
  transpose_w_kernel<<<dim3(32, 32), 256, 0, stream>>>(Wp, Wpt);

  // 2) projections: Q,K -> [b,h,t,64]; V -> [b,h,64,t]
  dim3 ggrid(16, 32);
  gemm_32x64_kernel<<<ggrid, 256, 0, stream>>>(xb, Wqt, nullptr, Qh, 1);
  gemm_32x64_kernel<<<ggrid, 256, 0, stream>>>(xb, Wkt, nullptr, Kh, 1);
  gemm_32x64_kernel<<<ggrid, 256, 0, stream>>>(xb, Wvt, nullptr, Vth, 2);

  // 3) fused softmax(QK^T/8)V -> ctx (overwrites xb; x no longer needed)
  flash_attn_kernel<<<dim3(64, 128), 32, 0, stream>>>(Qh, Kh, Vth, xb);

  // 4) out = ctx @ Wp  (f32 output)
  gemm_32x64_kernel<<<ggrid, 256, 0, stream>>>(xb, Wpt, out, nullptr, 0);
}
